// LSTMModel_31817117729004
// MI455X (gfx1250) — matrix operand order repaired and checked by reading, not compiled, not run on hardware
//
#include <hip/hip_runtime.h>
#include <hip/hip_bf16.h>

// ---------------- problem constants ----------------
#define BATCH    256
#define SEQ      512
#define IDIM     128
#define HDIM     512
#define GDIM     2048          // 4*HDIM
#define KDIM     640           // IDIM + HDIM
#define KTILES   20            // KDIM / 32
#define NTILES   128           // GDIM / 16
#define LSTRIDE  648           // LDS row stride in bf16 elems (multiple of 8 -> 16B aligned)
#define THREADS  512           // 16 wave32

typedef __attribute__((ext_vector_type(16))) __bf16        v16bf;
typedef __attribute__((ext_vector_type(8)))  float         v8f;
typedef __attribute__((ext_vector_type(4)))  unsigned int  u32x4;

union FragU { v16bf v; u32x4 q[2]; };

static __device__ __forceinline__ unsigned short f2bf(float f) {
    unsigned int u = __float_as_uint(f);
    u += 0x7FFFu + ((u >> 16) & 1u);       // round to nearest even
    return (unsigned short)(u >> 16);
}
static __device__ __forceinline__ float bf2f(unsigned short b) {
    return __uint_as_float(((unsigned int)b) << 16);
}
static __device__ __forceinline__ float sigmoidf_fast(float x) {
    return 1.0f / (1.0f + __expf(-x));
}
static __device__ __forceinline__ float tanhf_fast(float x) {
    return 2.0f / (1.0f + __expf(-2.0f * x)) - 1.0f;
}

// ---------------------------------------------------------------------------
// Kernel 1: pack [w_ih | w_hh] (f32) into bf16 B-fragments.
// Per 32x16 (KxN) tile: lane L holds N = nT*16 + L%16, h = L/16, 16 bf16 ordered
// K = kt*32 + 8h + v (v<8), K = kt*32 + 16 + 8h + (v-8) (v>=8).
// Storage: wpk[((nT*KTILES + kt)*32 + lane)*16 + v] -> 1KB contiguous per wave.
// Also biasC[n] = b_ih[n] + b_hh[n].
// ---------------------------------------------------------------------------
__global__ void lstm_pack_weights(const float* __restrict__ w_ih,
                                  const float* __restrict__ w_hh,
                                  const float* __restrict__ b_ih,
                                  const float* __restrict__ b_hh,
                                  unsigned short* __restrict__ wpk,
                                  float* __restrict__ biasC) {
    int tid = blockIdx.x * 256 + threadIdx.x;   // 128*20*32 = 81920 threads
    if (tid < NTILES * KTILES * 32) {
        int lane = tid & 31;
        int frag = tid >> 5;          // = nT*KTILES + kt
        int kt   = frag % KTILES;
        int nT   = frag / KTILES;
        int n    = nT * 16 + (lane & 15);
        unsigned short* dst = wpk + ((size_t)frag * 32 + lane) * 16;
        #pragma unroll
        for (int v = 0; v < 16; ++v) {
            int k = kt * 32 + (v < 8 ? 8 * (lane >> 4) + v : 16 + 8 * (lane >> 4) + (v - 8));
            float w = (k < IDIM) ? w_ih[(size_t)n * IDIM + k]
                                 : w_hh[(size_t)n * HDIM + (k - IDIM)];
            dst[v] = f2bf(w);
        }
    }
    if (tid < GDIM) biasC[tid] = b_ih[tid] + b_hh[tid];
}

// ---------------------------------------------------------------------------
// Kernel 2: persistent LSTM. One block = one 16-row batch tile, full sequence.
// 16 waves; wave w owns hidden units [32w, 32w+32): 8 accumulator tiles
// (4 gates x 2 hidden tiles, j = g*2 + lt). Cell state c stays in VGPRs for
// all 512 steps. Per kt: batch-load A + 8 B fragments (statically indexed,
// register-resident), then 8 WMMAs -> loads of the next kt overlap compute.
// ---------------------------------------------------------------------------
__global__ __launch_bounds__(THREADS, 1)
void lstm_persistent(const float* __restrict__ x,
                     const unsigned short* __restrict__ wpk,
                     const float* __restrict__ biasC,
                     const float* __restrict__ w_lin,
                     const float* __restrict__ b_lin,
                     float* __restrict__ out) {
    // act row r: cols [0,128) = x_t (bf16), cols [128,640) = h_{t-1} (bf16)
    __shared__ __align__(16) unsigned short act[16 * LSTRIDE];

    const int tid  = threadIdx.x;
    const int wave = tid >> 5;
    const int lane = tid & 31;
    const int b0   = blockIdx.x * 16;
    const int ht0  = wave * 2;          // first hidden tile owned by this wave
    const int ln15 = lane & 15;
    const int lhi  = lane >> 4;         // 0 / 1 half-wave

    // zero LDS (h region must start as 0)
    for (int i = tid; i < 16 * LSTRIDE; i += THREADS) act[i] = 0;
    __syncthreads();
    // stage x_0
    for (int i = tid; i < 16 * IDIM; i += THREADS) {
        int row = i >> 7, k = i & (IDIM - 1);
        act[row * LSTRIDE + k] = f2bf(x[((size_t)(b0 + row) * SEQ + 0) * IDIM + k]);
    }

    // combined biases for our 8 tiles (same for all M rows); j = g*2 + lt
    float bias[8];
    #pragma unroll
    for (int j = 0; j < 8; ++j)
        bias[j] = biasC[(j >> 1) * HDIM + (ht0 + (j & 1)) * 16 + ln15];

    // cell state in C/D fragment layout: cst[lt][r] = c(row r+8*lhi, col ln15)
    v8f cst[2];
    #pragma unroll
    for (int lt = 0; lt < 2; ++lt)
        #pragma unroll
        for (int r = 0; r < 8; ++r) cst[lt][r] = 0.0f;

    __syncthreads();

    for (int t = 0; t < SEQ; ++t) {
        // Launder a zero SGPR offset: keeps the loads rooted at the global
        // __restrict__ kernarg pointer (-> global_load, not flat_load) while
        // making them non-invariant across t (no LICM hoist / scratch spill).
        unsigned int lz = 0;
        asm volatile("" : "+s"(lz));
        const unsigned short* wp = wpk + lz;

        // init accumulators with bias; acc[j], j = g*2 + lt
        v8f acc[8];
        #pragma unroll
        for (int j = 0; j < 8; ++j)
            #pragma unroll
            for (int r = 0; r < 8; ++r) acc[j][r] = bias[j];

        // K loop, fully unrolled: 20 x (1 A-frag + 8 B-frags + 8 WMMA)
        #pragma unroll
        for (int kt = 0; kt < KTILES; ++kt) {
            const int k0 = kt * 32;
            // A fragment (16-bit A 16x32 layout): lane = row M=ln15,
            // halves K=[k0+8*lhi,+8) -> v0..3, K=[k0+16+8*lhi,+8) -> v4..7
            FragU fa;
            const unsigned short* ar = &act[ln15 * LSTRIDE + k0 + 8 * lhi];
            fa.q[0] = *(const u32x4*)(ar);
            fa.q[1] = *(const u32x4*)(ar + 16);

            // batch-load all 8 B fragments (1 KB/wave each, contiguous, NT)
            FragU fb[8];
            #pragma unroll
            for (int j = 0; j < 8; ++j) {
                const int nT = (j >> 1) * 32 + ht0 + (j & 1);
                const u32x4* bp = (const u32x4*)(wp +
                    ((size_t)(nT * KTILES + kt) * 32 + lane) * 16);
                fb[j].q[0] = __builtin_nontemporal_load(bp);
                fb[j].q[1] = __builtin_nontemporal_load(bp + 1);
            }

            #pragma unroll
            for (int j = 0; j < 8; ++j)
                acc[j] = __builtin_amdgcn_wmma_f32_16x16x32_bf16(
                    false, fa.v, false, fb[j].v, (short)0, acc[j],
                    false, false);
        }

        __syncthreads();   // all reads of act for step t complete

        // pointwise LSTM cell update; write h_t back to LDS (bf16)
        #pragma unroll
        for (int lt = 0; lt < 2; ++lt) {
            const int colbase = IDIM + (ht0 + lt) * 16 + ln15;
            #pragma unroll
            for (int r = 0; r < 8; ++r) {
                float iv = sigmoidf_fast(acc[0 + lt][r]);
                float fv = sigmoidf_fast(acc[2 + lt][r]);
                float gv = tanhf_fast   (acc[4 + lt][r]);
                float ov = sigmoidf_fast(acc[6 + lt][r]);
                float cv = fv * cst[lt][r] + iv * gv;
                cst[lt][r] = cv;
                float hv = ov * tanhf_fast(cv);
                int row = r + 8 * lhi;      // C/D layout: lanes 16-31 -> M+8
                act[row * LSTRIDE + colbase] = f2bf(hv);
            }
        }

        // stage x_{t+1}
        if (t + 1 < SEQ) {
            for (int i = tid; i < 16 * IDIM; i += THREADS) {
                int row = i >> 7, k = i & (IDIM - 1);
                act[row * LSTRIDE + k] =
                    f2bf(x[((size_t)(b0 + row) * SEQ + (t + 1)) * IDIM + k]);
            }
        }
        __syncthreads();   // h_t (and x_{t+1}) visible before next step
    }

    // final linear: wave w reduces batch row w (16 waves, 16 rows)
    {
        const int row = wave;
        float s = 0.0f;
        for (int j = lane; j < HDIM; j += 32)
            s += bf2f(act[row * LSTRIDE + IDIM + j]) * w_lin[j];
        #pragma unroll
        for (int off = 16; off > 0; off >>= 1) s += __shfl_xor(s, off, 32);
        if (lane == 0) out[b0 + row] = s + b_lin[0];
    }
}

// ---------------------------------------------------------------------------
extern "C" void kernel_launch(void* const* d_in, const int* in_sizes, int n_in,
                              void* d_out, int out_size, void* d_ws, size_t ws_size,
                              hipStream_t stream) {
    const float* x     = (const float*)d_in[0];
    const float* w_ih  = (const float*)d_in[1];
    const float* w_hh  = (const float*)d_in[2];
    const float* b_ih  = (const float*)d_in[3];
    const float* b_hh  = (const float*)d_in[4];
    const float* w_lin = (const float*)d_in[5];
    const float* b_lin = (const float*)d_in[6];
    float* out = (float*)d_out;

    unsigned short* wpk = (unsigned short*)d_ws;                       // 2.5 MB bf16
    float* biasC = (float*)((char*)d_ws + (size_t)GDIM * KDIM * 2);    // 8 KB f32

    // pack weights + combine biases (128*20*32 lanes -> 320 blocks of 256)
    lstm_pack_weights<<<320, 256, 0, stream>>>(w_ih, w_hh, b_ih, b_hh, wpk, biasC);

    // persistent LSTM: 16 batch tiles x 512 threads, full sequence per block
    lstm_persistent<<<BATCH / 16, THREADS, 0, stream>>>(x, wpk, biasC, w_lin,
                                                        b_lin, out);
}
